// TransformerBlock_57105885168168
// MI455X (gfx1250) — matrix operand order repaired and checked by reading, not compiled, not run on hardware
//
#include <hip/hip_runtime.h>
#include <hip/hip_bf16.h>
#include <math.h>

#define EMBED 1024
#define HEADS 16
#define HDIM 64
#define SEQ 2048
#define BATCH 2
#define MROWS (BATCH*SEQ)   /* 4096 */
#define MLPDIM 4096

typedef __bf16 bf16;
typedef __attribute__((ext_vector_type(16))) __bf16 bf16x16;
typedef __attribute__((ext_vector_type(8)))  __bf16 bf16x8;
typedef __attribute__((ext_vector_type(4)))  __bf16 bf16x4;
typedef __attribute__((ext_vector_type(8)))  float  f32x8;
typedef __attribute__((ext_vector_type(4)))  float  f32x4;

static __device__ __forceinline__ f32x8 wmma_bf16(bf16x16 a, bf16x16 b, f32x8 c) {
  return __builtin_amdgcn_wmma_f32_16x16x32_bf16(false, a, false, b, (short)0, c, false, false);
}

// Build a 16-element fragment from two 8-element (16B) LDS chunks.
static __device__ __forceinline__ bf16x16 frag_from2(const bf16* lo, const bf16* hi) {
  bf16x8 l = *(const bf16x8*)lo;
  bf16x8 h = *(const bf16x8*)hi;
  bf16x16 r;
#pragma unroll
  for (int i = 0; i < 8; ++i) { r[i] = l[i]; r[i+8] = h[i]; }
  return r;
}

// ---------------------------------------------------------------- small kernels
__global__ void silu_kernel(const float* __restrict__ in, float* __restrict__ out, int n) {
  int i = blockIdx.x * blockDim.x + threadIdx.x;
  if (i < n) { float x = in[i]; out[i] = x / (1.f + __expf(-x)); }
}

__global__ __launch_bounds__(256)
void modulation_kernel(const float* __restrict__ sc,
                       const float* __restrict__ scale_w, const float* __restrict__ scale_b,
                       const float* __restrict__ shift_w, const float* __restrict__ shift_b,
                       float* __restrict__ scale_out, float* __restrict__ shift_out) {
  int w    = blockIdx.x * 8 + (threadIdx.x >> 5);   // 4096 waves total
  int lane = threadIdx.x & 31;
  int sel  = w >> 11;          // 0 = shift, 1 = scale
  int rem  = w & 2047;
  int b    = rem >> 10;
  int col  = rem & 1023;
  const float* W = (sel ? scale_w : shift_w) + (size_t)col * EMBED;
  const float* s = sc + (size_t)b * EMBED;
  float acc = 0.f;
  for (int k = lane; k < EMBED; k += 32) acc += s[k] * W[k];
#pragma unroll
  for (int off = 16; off > 0; off >>= 1) acc += __shfl_xor(acc, off);
  if (lane == 0) {
    if (sel) {
      float v = acc + scale_b[col] + 1.f;
      scale_out[b * EMBED + col] = fminf(10.f, fmaxf(0.1f, v));
    } else {
      shift_out[b * EMBED + col] = acc + shift_b[col];
    }
  }
}

__global__ void cvt_kernel(const float* __restrict__ in, bf16* __restrict__ out) {
  size_t i = ((size_t)blockIdx.x * blockDim.x + threadIdx.x) * 8;
  f32x4 a = *(const f32x4*)(in + i);
  f32x4 b = *(const f32x4*)(in + i + 4);
  bf16x8 o;
#pragma unroll
  for (int e = 0; e < 4; ++e) { o[e] = (bf16)a[e]; o[e+4] = (bf16)b[e]; }
  *(bf16x8*)(out + i) = o;
}

// RMS-norm + adaLN modulation; fp32 in -> bf16 out. One block per (b,n) row.
__global__ __launch_bounds__(256)
void adaln_kernel(const float* __restrict__ x, const float* __restrict__ norm_w,
                  const float* __restrict__ scale, const float* __restrict__ shift,
                  bf16* __restrict__ out) {
  __shared__ float red[256];
  int row = blockIdx.x;
  int b   = row >> 11;
  int t   = threadIdx.x;
  const float* xr = x + (size_t)row * EMBED;
  f32x4 v = *(const f32x4*)(xr + t * 4);
  red[t] = v[0]*v[0] + v[1]*v[1] + v[2]*v[2] + v[3]*v[3];
  __syncthreads();
  for (int o = 128; o > 0; o >>= 1) { if (t < o) red[t] += red[t + o]; __syncthreads(); }
  float nrm  = fmaxf(sqrtf(red[0]) * 0.03125f, 1e-6f);  // ||x|| * C^-1/2, C=1024
  float rinv = 1.f / nrm;
  f32x4 nw = *(const f32x4*)(norm_w + t * 4);
  f32x4 s1 = *(const f32x4*)(scale + (size_t)b * EMBED + t * 4);
  f32x4 s0 = *(const f32x4*)(shift + (size_t)b * EMBED + t * 4);
  bf16x4 ov;
#pragma unroll
  for (int i = 0; i < 4; ++i) ov[i] = (bf16)(v[i] * rinv * nw[i] * s1[i] + s0[i]);
  *(bf16x4*)(out + (size_t)row * EMBED + t * 4) = ov;
}

// SwiGLU gate: h (Mx8192 bf16) -> silu(h[:, :4096]) * h[:, 4096:]
__global__ __launch_bounds__(256)
void gate_kernel(const bf16* __restrict__ h, bf16* __restrict__ out) {
  size_t e   = ((size_t)blockIdx.x * 256 + threadIdx.x) * 8;
  size_t row = e >> 12;
  size_t i   = e & 4095;
  bf16x8 x1 = *(const bf16x8*)(h + row * 8192 + i);
  bf16x8 x2 = *(const bf16x8*)(h + row * 8192 + 4096 + i);
  bf16x8 o;
#pragma unroll
  for (int k = 0; k < 8; ++k) {
    float a = (float)x1[k], b = (float)x2[k];
    o[k] = (bf16)((a / (1.f + __expf(-a))) * b);
  }
  *(bf16x8*)(out + row * 4096 + i) = o;
}

// ---------------------------------------------------------------- GEMM: O = A * W^T (+bias)(+res)
// A: MxK bf16 row-major.  W: NxK bf16 row-major.  Tiles 128x128x32, 256 threads (8 waves).
template<bool OUT_BF16, bool HAS_BIAS, bool HAS_RES>
__global__ __launch_bounds__(256)
void gemm_bf16_kernel(const bf16* __restrict__ A, const bf16* __restrict__ W,
                      const float* __restrict__ bias, const float* __restrict__ res,
                      void* __restrict__ outp, int M, int N, int K) {
  __shared__ bf16 lA[128 * 40];   // stride 40 halves (80B) -> conflict-free b128 reads
  __shared__ bf16 lB[32 * 136];   // K-major (transposed W tile), stride 136 halves
  const int t    = threadIdx.x;
  const int lane = t & 31;
  const int wave = t >> 5;
  const int wm   = wave & 3;      // 4 waves along M (32 rows each)
  const int wn   = wave >> 2;     // 2 waves along N (64 cols each)
  const int m0   = blockIdx.y * 128;
  const int n0   = blockIdx.x * 128;

  f32x8 acc[2][4];
#pragma unroll
  for (int s = 0; s < 2; ++s)
#pragma unroll
    for (int u = 0; u < 4; ++u)
#pragma unroll
      for (int r = 0; r < 8; ++r) acc[s][u][r] = 0.f;

  for (int k0 = 0; k0 < K; k0 += 32) {
    // A tile 128x32 (row-major)
#pragma unroll
    for (int i = 0; i < 2; ++i) {
      int idx = t + 256 * i;
      int ar = idx >> 2, as = idx & 3;
      bf16x8 v = *(const bf16x8*)(A + (size_t)(m0 + ar) * K + k0 + as * 8);
      *(bf16x8*)(&lA[ar * 40 + as * 8]) = v;
    }
    // W tile 128x32, stored transposed -> lB[k][n]
#pragma unroll
    for (int i = 0; i < 2; ++i) {
      int idx = t + 256 * i;
      int nr = idx >> 2, ks = idx & 3;
      bf16x8 v = *(const bf16x8*)(W + (size_t)(n0 + nr) * K + k0 + ks * 8);
#pragma unroll
      for (int e = 0; e < 8; ++e) lB[(ks * 8 + e) * 136 + nr] = v[e];
    }
    __syncthreads();
    bf16x16 af[2];
#pragma unroll
    for (int s = 0; s < 2; ++s) {
      int m   = wm * 32 + s * 16 + (lane & 15);
      int klo = (lane >> 4) * 8;             // ISA A layout: lanes 0-15 K 0-7/16-23, lanes 16-31 K 8-15/24-31
      af[s] = frag_from2(&lA[m * 40 + klo], &lA[m * 40 + klo + 16]);
    }
#pragma unroll
    for (int u = 0; u < 4; ++u) {
      int nb = wn * 64 + u * 16;
      bf16x16 bfv;
#pragma unroll
      for (int i = 0; i < 16; ++i) bfv[i] = lB[((i >> 3) * 16 + (lane >> 4) * 8 + (i & 7)) * 136 + nb + (lane & 15)];
#pragma unroll
      for (int s = 0; s < 2; ++s) acc[s][u] = wmma_bf16(af[s], bfv, acc[s][u]);
    }
    __syncthreads();
  }
  // epilogue: lane holds column (lane&15), rows 8*(lane>>4)+r
#pragma unroll
  for (int s = 0; s < 2; ++s) {
    int rowbase = m0 + wm * 32 + s * 16 + ((lane >> 4) << 3);
#pragma unroll
    for (int u = 0; u < 4; ++u) {
      int col = n0 + wn * 64 + u * 16 + (lane & 15);
      float bv = HAS_BIAS ? bias[col] : 0.f;
#pragma unroll
      for (int r = 0; r < 8; ++r) {
        size_t idx = (size_t)(rowbase + r) * N + col;
        float v = acc[s][u][r] + bv;
        if (HAS_RES) v += res[idx];
        if (OUT_BF16) ((bf16*)outp)[idx] = (bf16)v;
        else          ((float*)outp)[idx] = v;
      }
    }
  }
}

// ---------------------------------------------------------------- flash attention
// qkv: (B,N,3,H,D) bf16.  One block per (64-row tile, head, batch); 128 threads = 4 waves,
// each wave owns 16 query rows. Online softmax; masked scores = -1e30 (reproduces the
// reference's all-masked -> uniform fallback exactly).
__global__ __launch_bounds__(128)
void attention_kernel(const bf16* __restrict__ qkv, const int* __restrict__ amask,
                      bf16* __restrict__ outp) {
  __shared__ bf16 lK[64 * 72];        // K^T tile [d][j]; also Q staging [m][d]
  __shared__ bf16 lV[64 * 72];        // V tile [j][d]
  __shared__ bf16 lP[4 * 16 * 72];    // per-wave P staging
  const int t    = threadIdx.x;
  const int lane = t & 31;
  const int wave = t >> 5;
  const int mt   = blockIdx.x;        // 0..31
  const int h    = blockIdx.y;
  const int b    = blockIdx.z;
  const int row0 = mt * 64;
  const size_t base = (size_t)b * SEQ * 3 * EMBED + (size_t)h * HDIM;

  // stage Q tile (64x64) into lK, build per-wave A fragments
#pragma unroll
  for (int i = 0; i < 4; ++i) {
    int idx = t + 128 * i;            // 0..511
    int r = idx >> 3, ds = idx & 7;
    bf16x8 v = *(const bf16x8*)(qkv + base + (size_t)(row0 + r) * 3 * EMBED + ds * 8);
    *(bf16x8*)(&lK[r * 72 + ds * 8]) = v;
  }
  __syncthreads();
  bf16x16 aq[2];
  {
    int m = wave * 16 + (lane & 15);
#pragma unroll
    for (int ks = 0; ks < 2; ++ks) {
      int dlo = ks * 32 + (lane >> 4) * 8;
      aq[ks] = frag_from2(&lK[m * 72 + dlo], &lK[m * 72 + dlo + 16]);
    }
  }
  __syncthreads();

  float mrun[8], lrun[8];
  f32x8 o[4];
#pragma unroll
  for (int r = 0; r < 8; ++r) { mrun[r] = -1e30f; lrun[r] = 0.f; }
#pragma unroll
  for (int u = 0; u < 4; ++u)
#pragma unroll
    for (int r = 0; r < 8; ++r) o[u][r] = 0.f;

  for (int jt = 0; jt < SEQ; jt += 64) {
    // load K (transposed) and V tiles
#pragma unroll
    for (int i = 0; i < 4; ++i) {
      int idx = t + 128 * i;
      int r = idx >> 3, ds = idx & 7;
      const size_t rb = base + (size_t)(jt + r) * 3 * EMBED;
      bf16x8 kv = *(const bf16x8*)(qkv + rb + EMBED + ds * 8);
#pragma unroll
      for (int e = 0; e < 8; ++e) lK[(ds * 8 + e) * 72 + r] = kv[e];
      bf16x8 vv = *(const bf16x8*)(qkv + rb + 2 * EMBED + ds * 8);
      *(bf16x8*)(&lV[r * 72 + ds * 8]) = vv;
    }
    __syncthreads();

    // S = Q * K^T  (16x64 per wave)
    f32x8 sa[4];
#pragma unroll
    for (int u = 0; u < 4; ++u) {
#pragma unroll
      for (int r = 0; r < 8; ++r) sa[u][r] = 0.f;
#pragma unroll
      for (int ks = 0; ks < 2; ++ks) {
        bf16x16 bk;
#pragma unroll
        for (int i = 0; i < 16; ++i) bk[i] = lK[(ks * 32 + (i >> 3) * 16 + (lane >> 4) * 8 + (i & 7)) * 72 + u * 16 + (lane & 15)];
        sa[u] = wmma_bf16(aq[ks], bk, sa[u]);
      }
    }
    // scale + mask
    float sv[4][8];
#pragma unroll
    for (int u = 0; u < 4; ++u) {
      int j = jt + u * 16 + (lane & 15);
      bool ok = amask[b * SEQ + j] != 0;
#pragma unroll
      for (int r = 0; r < 8; ++r)
        sv[u][r] = ok ? sa[u][r] * 0.125f : -1e30f;
    }
    // online softmax row statistics (rows split across half-waves of 16 lanes)
#pragma unroll
    for (int r = 0; r < 8; ++r) {
      float mx = sv[0][r];
#pragma unroll
      for (int u = 1; u < 4; ++u) mx = fmaxf(mx, sv[u][r]);
#pragma unroll
      for (int off = 1; off < 16; off <<= 1) mx = fmaxf(mx, __shfl_xor(mx, off));
      float mn = fmaxf(mrun[r], mx);
      float al = __expf(mrun[r] - mn);
      mrun[r] = mn;
      float rs = 0.f;
#pragma unroll
      for (int u = 0; u < 4; ++u) {
        float p = __expf(sv[u][r] - mn);
        sv[u][r] = p;
        rs += p;
      }
#pragma unroll
      for (int off = 1; off < 16; off <<= 1) rs += __shfl_xor(rs, off);
      lrun[r] = lrun[r] * al + rs;
#pragma unroll
      for (int u = 0; u < 4; ++u) o[u][r] *= al;
    }
    // stage P (per-wave region, same-wave LDS ordering suffices)
    bf16* P = &lP[wave * 16 * 72];
#pragma unroll
    for (int u = 0; u < 4; ++u) {
      int j = u * 16 + (lane & 15);
#pragma unroll
      for (int r = 0; r < 8; ++r)
        P[((lane >> 4) * 8 + r) * 72 + j] = (bf16)sv[u][r];
    }
    // O += P * V
#pragma unroll
    for (int ks = 0; ks < 2; ++ks) {
      int m = lane & 15;
      int jlo = ks * 32 + (lane >> 4) * 8;
      bf16x16 ap = frag_from2(&P[m * 72 + jlo], &P[m * 72 + jlo + 16]);
#pragma unroll
      for (int u = 0; u < 4; ++u) {
        bf16x16 bv;
#pragma unroll
        for (int i = 0; i < 16; ++i) bv[i] = lV[(ks * 32 + (i >> 3) * 16 + (lane >> 4) * 8 + (i & 7)) * 72 + u * 16 + (lane & 15)];
        o[u] = wmma_bf16(ap, bv, o[u]);
      }
    }
    __syncthreads();
  }
  // finalize: divide by l, write (B,N,H*D) bf16
  float inv[8];
#pragma unroll
  for (int r = 0; r < 8; ++r) inv[r] = (lrun[r] > 0.f) ? 1.f / lrun[r] : 0.f;
  const size_t ob = (size_t)b * SEQ * EMBED + (size_t)h * HDIM;
#pragma unroll
  for (int u = 0; u < 4; ++u) {
    int d = u * 16 + (lane & 15);
#pragma unroll
    for (int r = 0; r < 8; ++r) {
      int row = row0 + wave * 16 + ((lane >> 4) << 3) + r;
      outp[ob + (size_t)row * EMBED + d] = (bf16)(o[u][r] * inv[r]);
    }
  }
}

// ---------------------------------------------------------------- launcher
extern "C" void kernel_launch(void* const* d_in, const int* in_sizes, int n_in,
                              void* d_out, int out_size, void* d_ws, size_t ws_size,
                              hipStream_t stream) {
  (void)in_sizes; (void)n_in; (void)out_size; (void)ws_size;
  const float* x       = (const float*)d_in[0];
  const float* c       = (const float*)d_in[1];
  const int*   amask   = (const int*)d_in[2];
  const float* norm_w  = (const float*)d_in[3];
  const float* scale_w = (const float*)d_in[4];
  const float* scale_b = (const float*)d_in[5];
  const float* shift_w = (const float*)d_in[6];
  const float* shift_b = (const float*)d_in[7];
  const float* qkv_w   = (const float*)d_in[8];
  const float* proj_w  = (const float*)d_in[9];
  const float* proj_b  = (const float*)d_in[10];
  const float* mlp_w1  = (const float*)d_in[11];
  const float* mlp_b1  = (const float*)d_in[12];
  const float* mlp_w2  = (const float*)d_in[13];
  const float* mlp_b2  = (const float*)d_in[14];
  float* out = (float*)d_out;

  char* ws = (char*)d_ws;
  size_t off = 0;
  auto take = [&](size_t b) -> char* {
    char* p = ws + off; off = (off + b + 255) & ~(size_t)255; return p;
  };
  float* sc      = (float*)take((size_t)BATCH * EMBED * 4);
  float* shiftv  = (float*)take((size_t)BATCH * EMBED * 4);
  float* scalev  = (float*)take((size_t)BATCH * EMBED * 4);
  bf16*  hbf     = (bf16*) take((size_t)MROWS * EMBED * 2);
  bf16*  qkvw_bf = (bf16*) take((size_t)3 * EMBED * EMBED * 2);
  bf16*  projw_bf= (bf16*) take((size_t)EMBED * EMBED * 2);
  bf16*  w1_bf   = (bf16*) take((size_t)2 * MLPDIM * EMBED * 2);
  bf16*  w2_bf   = (bf16*) take((size_t)EMBED * MLPDIM * 2);
  char*  big     = take((size_t)100663296);           // phase-shared scratch
  bf16*  qkvbuf  = (bf16*)big;                                          // 4096 x 3072
  bf16*  attnout = (bf16*)(big + (size_t)MROWS * 3 * EMBED * 2);        // 4096 x 1024
  bf16*  mlp1    = (bf16*)big;                                          // 4096 x 8192 (reuse)
  bf16*  gated   = (bf16*)(big + (size_t)MROWS * 2 * MLPDIM * 2);       // 4096 x 4096

  // 1) conditioning path (computed once, used by both adaLN applications)
  silu_kernel<<<8, 256, 0, stream>>>(c, sc, BATCH * EMBED);
  modulation_kernel<<<512, 256, 0, stream>>>(sc, scale_w, scale_b, shift_w, shift_b,
                                             scalev, shiftv);
  // 2) weight conversions fp32 -> bf16
  cvt_kernel<<<(3 * EMBED * EMBED) / 2048, 256, 0, stream>>>(qkv_w, qkvw_bf);
  cvt_kernel<<<(EMBED * EMBED) / 2048,     256, 0, stream>>>(proj_w, projw_bf);
  cvt_kernel<<<(2 * MLPDIM * EMBED) / 2048,256, 0, stream>>>(mlp_w1, w1_bf);
  cvt_kernel<<<(EMBED * MLPDIM) / 2048,    256, 0, stream>>>(mlp_w2, w2_bf);

  // 3) adaLN(x) -> hbf; qkv = hbf @ qkv_w^T
  adaln_kernel<<<MROWS, 256, 0, stream>>>(x, norm_w, scalev, shiftv, hbf);
  gemm_bf16_kernel<true, false, false><<<dim3(3 * EMBED / 128, MROWS / 128), 256, 0, stream>>>(
      hbf, qkvw_bf, nullptr, nullptr, qkvbuf, MROWS, 3 * EMBED, EMBED);

  // 4) attention
  attention_kernel<<<dim3(SEQ / 64, HEADS, BATCH), 128, 0, stream>>>(qkvbuf, amask, attnout);

  // 5) x_mid = x + attnout @ proj_w^T + proj_b   (into d_out)
  gemm_bf16_kernel<false, true, true><<<dim3(EMBED / 128, MROWS / 128), 256, 0, stream>>>(
      attnout, projw_bf, proj_b, x, out, MROWS, EMBED, EMBED);

  // 6) adaLN(x_mid) -> hbf; mlp1 = hbf @ w1^T + b1
  adaln_kernel<<<MROWS, 256, 0, stream>>>(out, norm_w, scalev, shiftv, hbf);
  gemm_bf16_kernel<true, true, false><<<dim3(2 * MLPDIM / 128, MROWS / 128), 256, 0, stream>>>(
      hbf, w1_bf, mlp_b1, nullptr, mlp1, MROWS, 2 * MLPDIM, EMBED);

  // 7) SwiGLU gate; out += gated @ w2^T + b2
  gate_kernel<<<(size_t)MROWS * MLPDIM / 2048, 256, 0, stream>>>(mlp1, gated);
  gemm_bf16_kernel<false, true, true><<<dim3(EMBED / 128, MROWS / 128), 256, 0, stream>>>(
      gated, w2_bf, mlp_b2, out, out, MROWS, EMBED, MLPDIM);
}
